// HeteroGNN_38001870635493
// MI455X (gfx1250) — matrix operand order repaired and checked by reading, not compiled, not run on hardware
//
#include <hip/hip_runtime.h>

typedef __attribute__((ext_vector_type(16))) _Float16 v16h;
typedef __attribute__((ext_vector_type(8)))  float    v8f;

#define DFEAT 64
#define HOUT  64
#define KTOT  128   // concat[aggmean(64), root(64)]

// ---------------- zero workspace (accumulators + counts) ----------------
__global__ void zero_kernel(float4* __restrict__ p, long n4) {
    long i      = (long)blockIdx.x * blockDim.x + threadIdx.x;
    long stride = (long)gridDim.x * blockDim.x;
    float4 z; z.x = 0.f; z.y = 0.f; z.z = 0.f; z.w = 0.f;
    for (; i < n4; i += stride) p[i] = z;
}

// ---------------- build transposed f16 combined weights ----------------
// wt[h*128 + k] = (k < 64 ? Wl[k][h] : Wr[k-64][h]) as f16, for both relations
__global__ void build_wt_kernel(const float* __restrict__ Wl_ur,
                                const float* __restrict__ Wr_ur,
                                const float* __restrict__ Wl_ru,
                                const float* __restrict__ Wr_ru,
                                _Float16* __restrict__ wt_ur,
                                _Float16* __restrict__ wt_ru) {
    int t = blockIdx.x * blockDim.x + threadIdx.x;   // 0..16383
    if (t >= 2 * HOUT * KTOT) return;
    int rel = t >> 13;           // 8192 entries per relation
    int i   = t & 8191;
    int h   = i >> 7;            // 0..63 output column
    int k   = i & 127;           // 0..127 concat-K
    const float* Wl = rel ? Wl_ru : Wl_ur;
    const float* Wr = rel ? Wr_ru : Wr_ur;
    float v = (k < DFEAT) ? Wl[k * HOUT + h] : Wr[(k - DFEAT) * HOUT + h];
    (rel ? wt_ru : wt_ur)[h * KTOT + k] = (_Float16)v;
}

// ---------------- edge scatter: 16 threads per edge, 4 floats each ----------------
__global__ void edge_scatter_kernel(const float* __restrict__ x_user,
                                    const float* __restrict__ x_res,
                                    const int*   __restrict__ src_u,
                                    const int*   __restrict__ dst_r,
                                    float* __restrict__ agg_r,
                                    float* __restrict__ agg_u,
                                    float* __restrict__ cnt_r,
                                    float* __restrict__ cnt_u,
                                    int E) {
    long gid = (long)blockIdx.x * blockDim.x + threadIdx.x;
    int  e    = (int)(gid >> 4);
    int  lane = (int)(gid & 15);
    if (e >= E) return;
    int s = src_u[e];
    int d = dst_r[e];
    int c = lane * 4;
    // forward: user feats -> resource accumulator
    float4 xu = *(const float4*)(x_user + (long)s * DFEAT + c);
    float* ar = agg_r + (long)d * DFEAT + c;
    atomicAdd(ar + 0, xu.x); atomicAdd(ar + 1, xu.y);
    atomicAdd(ar + 2, xu.z); atomicAdd(ar + 3, xu.w);
    // reverse: resource feats -> user accumulator
    float4 xr = *(const float4*)(x_res + (long)d * DFEAT + c);
    float* au = agg_u + (long)s * DFEAT + c;
    atomicAdd(au + 0, xr.x); atomicAdd(au + 1, xr.y);
    atomicAdd(au + 2, xr.z); atomicAdd(au + 3, xr.w);
    if (lane == 0) {
        atomicAdd(cnt_r + d, 1.0f);
        atomicAdd(cnt_u + s, 1.0f);
    }
}

// ---------------- fused SAGE GEMM: [nrows,128] x [128,64] via WMMA ----------------
// A = concat[agg/cnt , root], B = wt (f16, [n][k] row-major), bias in C init, ReLU epilogue.
// One wave per 16-row tile; 4 K-steps x 4 N-tiles = 16 v_wmma_f32_16x16x32_f16.
__global__ void sage_gemm_kernel(const float*    __restrict__ agg,
                                 const float*    __restrict__ cnt,
                                 const float*    __restrict__ xroot,
                                 const _Float16* __restrict__ wt,
                                 const float*    __restrict__ bias,
                                 float*          __restrict__ out,
                                 int nrows) {
    int wave = (int)(((long)blockIdx.x * blockDim.x + threadIdx.x) >> 5);
    int lane = threadIdx.x & 31;
    int ntiles = (nrows + 15) >> 4;
    if (wave >= ntiles) return;

    int row0  = wave << 4;
    int lr    = lane & 15;     // row (A) / column (B,C,D) within tile
    int khalf = lane >> 4;     // lane-group select
    int r  = row0 + lr;
    int rc = (r < nrows) ? r : (nrows - 1);   // clamp for ragged tail reads
    float inv = 1.0f / fmaxf(cnt[rc], 1.0f);  // zero-degree clamp (mean semantics)

    // accumulators carry the bias (column-only dependence matches C layout)
    v8f c[4];
#pragma unroll
    for (int nt = 0; nt < 4; ++nt) {
        float b = bias[nt * 16 + lr];
        c[nt] = (v8f){b, b, b, b, b, b, b, b};
    }

#pragma unroll
    for (int kt = 0; kt < 4; ++kt) {
        // K 0..63 from mean-aggr (scaled), K 64..127 from root feats
        const float* srow = (kt < 2) ? (agg   + (long)rc * DFEAT + (kt & 1) * 32)
                                     : (xroot + (long)rc * DFEAT + (kt & 1) * 32);
        float sc = (kt < 2) ? inv : 1.0f;
        int  k0  = khalf * 16;
        // A 16x32 f16 layout: lanes 0-15 hold K 0..15, lanes 16-31 hold K 16..31
        v16h a;
#pragma unroll
        for (int i = 0; i < 8; ++i) a[i]     = (_Float16)(srow[k0 + i]      * sc);
#pragma unroll
        for (int i = 0; i < 8; ++i) a[8 + i] = (_Float16)(srow[8 + k0 + i] * sc);

#pragma unroll
        for (int nt = 0; nt < 4; ++nt) {
            // B 32x16 f16: lane holds K = khalf*16 + 0..15 of column (nt*16+lr),
            // contiguous 32B in the [n][k] transposed weight buffer
            const v16h* bp = (const v16h*)(wt + (long)(nt * 16 + lr) * KTOT
                                              + kt * 32 + khalf * 16);
            v16h bf = *bp;
            c[nt] = __builtin_amdgcn_wmma_f32_16x16x32_f16(
                        false, a, false, bf, (short)0, c[nt], false, false);
        }
    }

    // epilogue: C/D layout -> out rows row0 + j + khalf*8, column nt*16+lr; ReLU
#pragma unroll
    for (int nt = 0; nt < 4; ++nt) {
#pragma unroll
        for (int j = 0; j < 8; ++j) {
            int grow = row0 + j + khalf * 8;
            if (grow < nrows) {
                float v = c[nt][j];
                out[(long)grow * HOUT + nt * 16 + lr] = (v > 0.0f) ? v : 0.0f;
            }
        }
    }
}

extern "C" void kernel_launch(void* const* d_in, const int* in_sizes, int n_in,
                              void* d_out, int out_size, void* d_ws, size_t ws_size,
                              hipStream_t stream) {
    const float* x_user = (const float*)d_in[0];
    const float* x_res  = (const float*)d_in[1];
    const int*   src_u  = (const int*)d_in[2];
    const int*   dst_r  = (const int*)d_in[3];
    const float* Wl_ur  = (const float*)d_in[4];
    const float* bl_ur  = (const float*)d_in[5];
    const float* Wr_ur  = (const float*)d_in[6];
    const float* Wl_ru  = (const float*)d_in[7];
    const float* bl_ru  = (const float*)d_in[8];
    const float* Wr_ru  = (const float*)d_in[9];

    long Nu = in_sizes[0] / DFEAT;   // 200000
    long Nr = in_sizes[1] / DFEAT;   // 50000
    int  E  = in_sizes[2];           // 2000000

    // workspace layout (all 32B-aligned given typical 256B base alignment)
    float* ws     = (float*)d_ws;
    float* agg_r  = ws;
    float* agg_u  = agg_r + Nr * DFEAT;
    float* cnt_r  = agg_u + Nu * DFEAT;
    float* cnt_u  = cnt_r + Nr;
    _Float16* wt_ur = (_Float16*)(cnt_u + Nu);
    _Float16* wt_ru = wt_ur + HOUT * KTOT;

    float* out_res  = (float*)d_out;            // [Nr, 64]
    float* out_user = out_res + Nr * HOUT;      // [Nu, 64]

    // 1) zero accumulators + counts
    long nzero4 = (Nr * DFEAT + Nu * DFEAT + Nr + Nu) / 4;
    zero_kernel<<<4096, 256, 0, stream>>>((float4*)agg_r, nzero4);

    // 2) build f16 transposed combined weights (both relations)
    build_wt_kernel<<<(2 * HOUT * KTOT + 255) / 256, 256, 0, stream>>>(
        Wl_ur, Wr_ur, Wl_ru, Wr_ru, wt_ur, wt_ru);

    // 3) edge scatter (both directions in one pass over edges)
    long total = (long)E * 16;
    int  blocks = (int)((total + 255) / 256);
    edge_scatter_kernel<<<blocks, 256, 0, stream>>>(
        x_user, x_res, src_u, dst_r, agg_r, agg_u, cnt_r, cnt_u, E);

    // 4) fused mean + GEMM + bias + ReLU, per node type
    {
        int ntiles = (int)((Nr + 15) / 16);
        int gb = (ntiles * 32 + 255) / 256;
        sage_gemm_kernel<<<gb, 256, 0, stream>>>(agg_r, cnt_r, x_res, wt_ur,
                                                 bl_ur, out_res, (int)Nr);
    }
    {
        int ntiles = (int)((Nu + 15) / 16);
        int gb = (ntiles * 32 + 255) / 256;
        sage_gemm_kernel<<<gb, 256, 0, stream>>>(agg_u, cnt_u, x_user, wt_ru,
                                                 bl_ru, out_user, (int)Nu);
    }
}
